// SparseSwiGLUFFN_36009005809979
// MI455X (gfx1250) — matrix operand order repaired and checked by reading, not compiled, not run on hardware
//
#include <hip/hip_runtime.h>

typedef __bf16 bf16_t;
typedef __attribute__((ext_vector_type(16))) __bf16 v16bf;
typedef __attribute__((ext_vector_type(8)))  __bf16 v8bf;
typedef __attribute__((ext_vector_type(8)))  float  v8f;
typedef __attribute__((ext_vector_type(4)))  int    v4i;

#define N_TOK   8192
#define D_MODEL 1024
#define D_FFN   4096
#define TOPK    512

#define TM 128          // rows per block tile
#define TN 64           // cols per block tile
#define TK 64           // k per LDS stage (two WMMA K-steps)
#define LDA 72          // padded LDS row stride (bf16): 144B, 16B-aligned
#define LDB 72

// ---------- CDNA5 async global->LDS copy (guarded; falls back to plain copy) ----
#if defined(__has_builtin)
#if __has_builtin(__builtin_amdgcn_global_load_async_to_lds_b128) && \
    __has_builtin(__builtin_amdgcn_s_wait_asynccnt)
#define HAS_ASYNC_LDS 1
#endif
#endif

typedef __attribute__((address_space(1))) v4i* gv4i_p;   // int4 __device__ *
typedef __attribute__((address_space(3))) v4i* lv4i_p;   // int4 __shared__ *

__device__ __forceinline__ void async_copy16(const void* g, void* l) {
#ifdef HAS_ASYNC_LDS
    // generic LDS pointer low 32 bits == LDS byte offset
    __builtin_amdgcn_global_load_async_to_lds_b128(
        (gv4i_p)(unsigned long long)g,
        (lv4i_p)(unsigned int)(unsigned long long)l, 0, 0);
#else
    *(v8bf*)l = *(const v8bf*)g;
#endif
}
__device__ __forceinline__ void async_wait_all() {
#ifdef HAS_ASYNC_LDS
    __builtin_amdgcn_s_wait_asynccnt(0);
#endif
}
__device__ __forceinline__ float fast_rcp(float x) {
#if defined(__has_builtin)
#if __has_builtin(__builtin_amdgcn_rcpf)
    return __builtin_amdgcn_rcpf(x);
#else
    return 1.0f / x;
#endif
#else
    return 1.0f / x;
#endif
}

// ---------------- fp32 -> bf16 conversion ----------------
__global__ void cvt_f32_bf16(const float* __restrict__ in, bf16_t* __restrict__ out, int n) {
    int i = blockIdx.x * blockDim.x + threadIdx.x;
    int stride = gridDim.x * blockDim.x;
    for (; i < n; i += stride) out[i] = (bf16_t)in[i];
}

// ---------------- per-row index histogram ----------------
// Z_dense[n,f] = count[n,f] * silu(G[n,f]) * U[n,f]
__global__ __launch_bounds__(256) void count_kernel(const int* __restrict__ idx,
                                                    unsigned char* __restrict__ cnt) {
    __shared__ unsigned int hist[D_FFN];   // 16 KB of 320 KB WGP LDS
    const int row = blockIdx.x;
    for (int f = threadIdx.x; f < D_FFN; f += 256) hist[f] = 0u;
    __syncthreads();
    const int* r = idx + (size_t)row * TOPK;
    for (int j = threadIdx.x; j < TOPK; j += 256) atomicAdd(&hist[r[j]], 1u);
    __syncthreads();
    unsigned char* o = cnt + (size_t)row * D_FFN;
    for (int f = threadIdx.x; f < D_FFN; f += 256) {
        unsigned int c = hist[f];
        o[f] = (unsigned char)(c > 255u ? 255u : c);
    }
}

// ---------------- WMMA fragment loaders (ISA VGPR layouts) ----------------
// A 16x32 bf16: lane<16 -> row M=lane,    elems[0..7]=K0..7,  [8..15]=K16..23
//               lane>=16 -> row M=lane-16, elems[0..7]=K8..15, [8..15]=K24..31
__device__ __forceinline__ v16bf load_a_frag(const bf16_t* As, int mr, int ko, int lane) {
    const int half = lane >> 4, lr = lane & 15;
    const bf16_t* base = As + (size_t)(mr + lr) * LDA + ko;
    v8bf lo = *(const v8bf*)(base + half * 8);
    v8bf hi = *(const v8bf*)(base + 16 + half * 8);
    return __builtin_shufflevector(lo, hi, 0,1,2,3,4,5,6,7,8,9,10,11,12,13,14,15);
}
// B 32x16 bf16 (staged transposed in LDS as [n][k]):
//   lane<16 -> col N=lane, elems[0..7]=K0..7, [8..15]=K16..23 ; lane>=16 -> col N=lane-16, elems[0..7]=K8..15, [8..15]=K24..31
__device__ __forceinline__ v16bf load_b_frag(const bf16_t* Bs, int nc, int ko, int lane) {
    const int half = lane >> 4, lr = lane & 15;
    const bf16_t* base = Bs + (size_t)(nc + lr) * LDB + ko;
    v8bf lo = *(const v8bf*)(base + half * 8);
    v8bf hi = *(const v8bf*)(base + 16 + half * 8);
    return __builtin_shufflevector(lo, hi, 0,1,2,3,4,5,6,7,8,9,10,11,12,13,14,15);
}

// ---------------- GEMM1: fused gate+up + count*SwiGLU epilogue ----------------
__global__ __launch_bounds__(256) void gemm_gate_up(
    const bf16_t* __restrict__ xb,          // [N_TOK][D_MODEL]
    const bf16_t* __restrict__ wg,          // [D_MODEL][D_FFN]
    const bf16_t* __restrict__ wu,          // [D_MODEL][D_FFN]
    const unsigned char* __restrict__ cnt,  // [N_TOK][D_FFN]
    bf16_t* __restrict__ zb)                // [N_TOK][D_FFN]
{
    __shared__ alignas(16) bf16_t As[TM * LDA];
    __shared__ alignas(16) bf16_t Bg[TN * LDB];
    __shared__ alignas(16) bf16_t Bu[TN * LDB];

    const int tid  = threadIdx.x;
    const int lane = tid & 31;
    const int wid  = tid >> 5;      // 8 waves
    const int wm   = wid & 3;       // 4 row-groups of 32
    const int wn   = wid >> 2;      // 2 col-groups of 32
    const int m0   = blockIdx.y * TM;
    const int n0   = blockIdx.x * TN;

    v8f accG[2][2] = {};
    v8f accU[2][2] = {};

    for (int k0 = 0; k0 < D_MODEL; k0 += TK) {
        // A tile 128x64 via async global->LDS (16B per lane, no transpose needed)
        #pragma unroll
        for (int i = 0; i < 4; ++i) {
            int c   = tid + i * 256;       // 0..1023
            int row = c >> 3;              // 0..127
            int kp  = (c & 7) * 8;         // 0..56
            async_copy16(xb + (size_t)(m0 + row) * D_MODEL + k0 + kp,
                         As + row * LDA + kp);
        }
        // B tiles 64x64: coalesced row-major global read, transposed LDS store
        #pragma unroll
        for (int i = 0; i < 2; ++i) {
            int c  = tid + i * 256;        // 0..511
            int kk = c >> 3;               // 0..63
            int np = (c & 7) * 8;          // 0..56
            v8bf gv = *(const v8bf*)(wg + (size_t)(k0 + kk) * D_FFN + n0 + np);
            v8bf uv = *(const v8bf*)(wu + (size_t)(k0 + kk) * D_FFN + n0 + np);
            #pragma unroll
            for (int j = 0; j < 8; ++j) {
                Bg[(np + j) * LDB + kk] = gv[j];
                Bu[(np + j) * LDB + kk] = uv[j];
            }
        }
        // prefetch next k-stage of the streamed weights into cache
        if (k0 + TK < D_MODEL) {
            int kk = tid >> 3, np = (tid & 7) * 8;
            __builtin_prefetch(wg + (size_t)(k0 + TK + kk) * D_FFN + n0 + np, 0, 0);
            __builtin_prefetch(wu + (size_t)(k0 + TK + kk) * D_FFN + n0 + np, 0, 0);
        }
        async_wait_all();
        __syncthreads();

        #pragma unroll
        for (int ks = 0; ks < 2; ++ks) {
            const int ko = ks * 32;
            v16bf afrag[2], bgfrag[2], bufrag[2];
            #pragma unroll
            for (int i = 0; i < 2; ++i) afrag[i] = load_a_frag(As, wm * 32 + i * 16, ko, lane);
            #pragma unroll
            for (int j = 0; j < 2; ++j) {
                bgfrag[j] = load_b_frag(Bg, wn * 32 + j * 16, ko, lane);
                bufrag[j] = load_b_frag(Bu, wn * 32 + j * 16, ko, lane);
            }
            #pragma unroll
            for (int i = 0; i < 2; ++i) {
                #pragma unroll
                for (int j = 0; j < 2; ++j) {
                    accG[i][j] = __builtin_amdgcn_wmma_f32_16x16x32_bf16(
                        false, afrag[i], false, bgfrag[j], (short)0, accG[i][j], false, false);
                    accU[i][j] = __builtin_amdgcn_wmma_f32_16x16x32_bf16(
                        false, afrag[i], false, bufrag[j], (short)0, accU[i][j], false, false);
                }
            }
        }
        __syncthreads();
    }

    // Epilogue: D layout is VGPR v -> M=v (lane<16) / M=v+8 (lane>=16), N=lane&15
    const int half = lane >> 4, lr = lane & 15;
    #pragma unroll
    for (int i = 0; i < 2; ++i) {
        #pragma unroll
        for (int j = 0; j < 2; ++j) {
            const int col   = n0 + wn * 32 + j * 16 + lr;
            const int rbase = m0 + wm * 32 + i * 16 + half * 8;
            #pragma unroll
            for (int v = 0; v < 8; ++v) {
                const int row = rbase + v;
                float g = accG[i][j][v];
                float u = accU[i][j][v];
                float c = (float)cnt[(size_t)row * D_FFN + col];
                float s = g * fast_rcp(1.0f + __expf(-g));   // silu
                zb[(size_t)row * D_FFN + col] = (bf16_t)(c * s * u);
            }
        }
    }
}

// ---------------- GEMM2: out = Z(bf16) @ w_down(bf16), f32 out ----------------
__global__ __launch_bounds__(256) void gemm_down(
    const bf16_t* __restrict__ zb,   // [N_TOK][D_FFN]
    const bf16_t* __restrict__ wd,   // [D_FFN][D_MODEL]
    float* __restrict__ out)         // [N_TOK][D_MODEL]
{
    __shared__ alignas(16) bf16_t As[TM * LDA];
    __shared__ alignas(16) bf16_t Bs[TN * LDB];

    const int tid  = threadIdx.x;
    const int lane = tid & 31;
    const int wid  = tid >> 5;
    const int wm   = wid & 3;
    const int wn   = wid >> 2;
    const int m0   = blockIdx.y * TM;
    const int n0   = blockIdx.x * TN;

    v8f acc[2][2] = {};

    for (int k0 = 0; k0 < D_FFN; k0 += TK) {
        #pragma unroll
        for (int i = 0; i < 4; ++i) {
            int c   = tid + i * 256;
            int row = c >> 3;
            int kp  = (c & 7) * 8;
            async_copy16(zb + (size_t)(m0 + row) * D_FFN + k0 + kp,
                         As + row * LDA + kp);
        }
        #pragma unroll
        for (int i = 0; i < 2; ++i) {
            int c  = tid + i * 256;
            int kk = c >> 3;
            int np = (c & 7) * 8;
            v8bf wv = *(const v8bf*)(wd + (size_t)(k0 + kk) * D_MODEL + n0 + np);
            #pragma unroll
            for (int j = 0; j < 8; ++j) Bs[(np + j) * LDB + kk] = wv[j];
        }
        if (k0 + TK < D_FFN) {
            int kk = tid >> 3, np = (tid & 7) * 8;
            __builtin_prefetch(wd + (size_t)(k0 + TK + kk) * D_MODEL + n0 + np, 0, 0);
        }
        async_wait_all();
        __syncthreads();

        #pragma unroll
        for (int ks = 0; ks < 2; ++ks) {
            const int ko = ks * 32;
            v16bf afrag[2], bfrag[2];
            #pragma unroll
            for (int i = 0; i < 2; ++i) afrag[i] = load_a_frag(As, wm * 32 + i * 16, ko, lane);
            #pragma unroll
            for (int j = 0; j < 2; ++j) bfrag[j] = load_b_frag(Bs, wn * 32 + j * 16, ko, lane);
            #pragma unroll
            for (int i = 0; i < 2; ++i) {
                #pragma unroll
                for (int j = 0; j < 2; ++j) {
                    acc[i][j] = __builtin_amdgcn_wmma_f32_16x16x32_bf16(
                        false, afrag[i], false, bfrag[j], (short)0, acc[i][j], false, false);
                }
            }
        }
        __syncthreads();
    }

    const int half = lane >> 4, lr = lane & 15;
    #pragma unroll
    for (int i = 0; i < 2; ++i) {
        #pragma unroll
        for (int j = 0; j < 2; ++j) {
            const int col   = n0 + wn * 32 + j * 16 + lr;
            const int rbase = m0 + wm * 32 + i * 16 + half * 8;
            #pragma unroll
            for (int v = 0; v < 8; ++v) {
                out[(size_t)(rbase + v) * D_MODEL + col] = acc[i][j][v];
            }
        }
    }
}

// ---------------- launcher ----------------
extern "C" void kernel_launch(void* const* d_in, const int* in_sizes, int n_in,
                              void* d_out, int out_size, void* d_ws, size_t ws_size,
                              hipStream_t stream) {
    const float* x      = (const float*)d_in[0];
    const int*   topk   = (const int*)  d_in[1];
    const float* w_gate = (const float*)d_in[2];
    const float* w_up   = (const float*)d_in[3];
    const float* w_down = (const float*)d_in[4];
    float* out = (float*)d_out;

    // Workspace layout (bytes): xb 16M | wgb 8M | wub 8M | wdb 8M | cnt 32M | zb 64M
    bf16_t* xb  = (bf16_t*)d_ws;
    bf16_t* wgb = xb  + (size_t)N_TOK  * D_MODEL;
    bf16_t* wub = wgb + (size_t)D_MODEL * D_FFN;
    bf16_t* wdb = wub + (size_t)D_MODEL * D_FFN;
    unsigned char* cnt = (unsigned char*)(wdb + (size_t)D_FFN * D_MODEL);
    bf16_t* zb  = (bf16_t*)(cnt + (size_t)N_TOK * D_FFN);

    cvt_f32_bf16<<<2048, 256, 0, stream>>>(x,      xb,  N_TOK * D_MODEL);
    cvt_f32_bf16<<<2048, 256, 0, stream>>>(w_gate, wgb, D_MODEL * D_FFN);
    cvt_f32_bf16<<<2048, 256, 0, stream>>>(w_up,   wub, D_MODEL * D_FFN);
    cvt_f32_bf16<<<2048, 256, 0, stream>>>(w_down, wdb, D_FFN * D_MODEL);

    count_kernel<<<N_TOK, 256, 0, stream>>>(topk, cnt);

    gemm_gate_up<<<dim3(D_FFN / TN, N_TOK / TM), 256, 0, stream>>>(xb, wgb, wub, cnt, zb);
    gemm_down  <<<dim3(D_MODEL / TN, N_TOK / TM), 256, 0, stream>>>(zb, wdb, out);
}
